// MultiHeadAttention_20658792694404
// MI455X (gfx1250) — matrix operand order repaired and checked by reading, not compiled, not run on hardware
//
#include <hip/hip_runtime.h>
#include <hip/hip_bf16.h>

typedef __attribute__((ext_vector_type(16))) _Float16 v16h;
typedef __attribute__((ext_vector_type(8)))  _Float16 v8h;
typedef __attribute__((ext_vector_type(8)))  float    v8f;

#define D_MODEL  1024
#define N_HEADS  16
#define HEAD_DIM 64
#define BATCH    2
#define SEQ      2048
#define M_TOTAL  (BATCH * SEQ)   // 4096

// ---------------------------------------------------------------------------
// fp32 -> f16 conversion (grid-stride)
// ---------------------------------------------------------------------------
__global__ void f32_to_f16_kernel(const float* __restrict__ src,
                                  _Float16* __restrict__ dst, int n) {
    int i = blockIdx.x * blockDim.x + threadIdx.x;
    int stride = gridDim.x * blockDim.x;
    for (; i < n; i += stride) dst[i] = (_Float16)src[i];
}

// ---------------------------------------------------------------------------
// Tiled WMMA GEMM: C[M,N] = A[M,K](f16,row-major) * B[K,N](f16,row-major)+bias
// 128 threads = 4 waves; 128x64 output tile; wave w owns rows 32w..32w+31
// (two 16-row m-tiles sharing each B fragment).
// MODE 0: fp32 row-major output.  MODE 1: f16 output scattered to [b,h,l,d].
// ---------------------------------------------------------------------------
template <int MODE>
__global__ __launch_bounds__(128)
void gemm_wmma(const _Float16* __restrict__ A, const _Float16* __restrict__ B,
               const float* __restrict__ bias, float* __restrict__ outF,
               _Float16* __restrict__ outH, int M, int N, int K) {
    __shared__ __align__(16) _Float16 As[128][40];   // [m][k], 80B pitch
    __shared__ __align__(16) _Float16 Bt[64][40];    // [n][k] transposed

    const int tid  = threadIdx.x;
    const int lane = tid & 31;
    const int wave = tid >> 5;
    const int m0 = blockIdx.y * 128;
    const int n0 = blockIdx.x * 64;

    // fragment lane geometry (per CDNA5 WMMA VGPR layouts)
    const int am  = lane & 15;             // A: row within 16
    const int kb  = (lane >> 4) << 3;      // A: K sub-base (0/8)
    const int bn  = lane & 15;             // B: column
    const int kb2 = (lane >> 4) << 4;      // 0/16, unused (B fragments use kb)
    const int mb  = (lane >> 4) << 3;      // C/D: row base (0/8)

    v8f acc0[4] = {};
    v8f acc1[4] = {};

    for (int k0 = 0; k0 < K; k0 += 32) {
        __syncthreads();
        // stage A tile 128x32: 512 8-elem chunks, 4 per thread (b128 ld/st)
        #pragma unroll
        for (int t = 0; t < 4; ++t) {
            int c = tid + t * 128;
            int mm = c >> 2, k8 = (c & 3) * 8;
            *(v8h*)&As[mm][k8] =
                *(const v8h*)&A[(size_t)(m0 + mm) * K + k0 + k8];
        }
        // stage B tile 32x64 transposed: 256 chunks, 2 per thread
        #pragma unroll
        for (int t = 0; t < 2; ++t) {
            int c = tid + t * 128;
            int kk = c >> 3, n8 = (c & 7) * 8;
            v8h v = *(const v8h*)&B[(size_t)(k0 + kk) * N + n0 + n8];
            #pragma unroll
            for (int j = 0; j < 8; ++j) Bt[n8 + j][kk] = v[j];
        }
        // prefetch next k-step tiles into cache while we compute
        if (k0 + 32 < K) {
            int cA = tid & 511;
            __builtin_prefetch(&A[(size_t)(m0 + (cA >> 2)) * K + k0 + 32], 0, 1);
            __builtin_prefetch(&B[(size_t)(k0 + 32 + (tid >> 3)) * N + n0], 0, 1);
        }
        __syncthreads();

        // A fragments (two 16-row tiles) for this wave
        v16h a0, a1;
        {
            const _Float16* ar0 = &As[wave * 32 + am][0];
            const _Float16* ar1 = &As[wave * 32 + 16 + am][0];
            #pragma unroll
            for (int i = 0; i < 8; ++i) {
                a0[i]     = ar0[kb + i];
                a0[8 + i] = ar0[16 + kb + i];
                a1[i]     = ar1[kb + i];
                a1[8 + i] = ar1[16 + kb + i];
            }
        }
        #pragma unroll
        for (int nt = 0; nt < 4; ++nt) {
            v16h b;
            const _Float16* br = &Bt[nt * 16 + bn][0];
            #pragma unroll
            for (int i = 0; i < 16; ++i) b[i] = br[(i < 8 ? kb : 16 + kb) + (i & 7)];
            acc0[nt] = __builtin_amdgcn_wmma_f32_16x16x32_f16(
                false, a0, false, b, (short)0, acc0[nt], false, false);
            acc1[nt] = __builtin_amdgcn_wmma_f32_16x16x32_f16(
                false, a1, false, b, (short)0, acc1[nt], false, false);
        }
    }

    // epilogue: two 16-row tiles
    #pragma unroll
    for (int mt = 0; mt < 2; ++mt) {
        #pragma unroll
        for (int nt = 0; nt < 4; ++nt) {
            int n = n0 + nt * 16 + bn;
            float bv = bias[n];
            #pragma unroll
            for (int r = 0; r < 8; ++r) {
                int m = m0 + wave * 32 + mt * 16 + mb + r;
                float v = (mt == 0 ? acc0[nt][r] : acc1[nt][r]) + bv;
                if (MODE == 0) {
                    outF[(size_t)m * N + n] = v;
                } else {
                    int b_ = m >> 11, l_ = m & (SEQ - 1);
                    int h_ = n >> 6,  d_ = n & (HEAD_DIM - 1);
                    outH[(((size_t)(b_ * N_HEADS + h_) * SEQ) + l_) * HEAD_DIM
                         + d_] = (_Float16)v;
                }
            }
        }
    }
}

// ---------------------------------------------------------------------------
// Flash attention: one workgroup = 64 query rows of one (b,h); 4 waves, each
// wave owns a 16-row Q block. Causal, online softmax, fp32 accumulation.
// K tile is staged by the gfx1250 async Global->LDS engine (ASYNCcnt);
// V tile needs a transpose so it goes through VGPRs.
// ---------------------------------------------------------------------------
__global__ __launch_bounds__(128)
void attn_wmma(const _Float16* __restrict__ Qh, const _Float16* __restrict__ Kh,
               const _Float16* __restrict__ Vh, _Float16* __restrict__ ctx) {
    __shared__ __align__(16) _Float16 Kt[64][72];      // [key][dim], 144B pitch
    __shared__ __align__(16) _Float16 Vt[64][72];      // [dim][key] transposed
    __shared__ __align__(16) _Float16 Pt[4][16][72];   // per-wave P tile

    const int tid  = threadIdx.x;
    const int lane = tid & 31;
    const int wave = tid >> 5;
    const int bh   = blockIdx.y;           // 0..31  (b*16 + h)
    const int qt0  = blockIdx.x * 64;
    const int q0   = qt0 + wave * 16;

    const _Float16* Qb = Qh + (size_t)bh * SEQ * HEAD_DIM;
    const _Float16* Kb = Kh + (size_t)bh * SEQ * HEAD_DIM;
    const _Float16* Vb = Vh + (size_t)bh * SEQ * HEAD_DIM;

    const int am  = lane & 15;
    const int kb  = (lane >> 4) << 3;
    const int bn  = lane & 15;
    const int kb2 = (lane >> 4) << 4;
    const int mb  = (lane >> 4) << 3;

    // Q fragments: contraction over 64 dims -> two 16x32 A fragments
    v16h aq0, aq1;
    {
        const _Float16* qr = Qb + (size_t)(q0 + am) * HEAD_DIM;
        #pragma unroll
        for (int i = 0; i < 8; ++i) {
            aq0[i]     = qr[kb + i];
            aq0[8 + i] = qr[16 + kb + i];
            aq1[i]     = qr[32 + kb + i];
            aq1[8 + i] = qr[48 + kb + i];
        }
    }

    v8f o[4] = {};
    float mrow[8], lrow[8];
    #pragma unroll
    for (int r = 0; r < 8; ++r) { mrow[r] = -1e30f; lrow[r] = 0.f; }
    const float scale = 0.125f;            // 1/sqrt(64)

    for (int k0 = 0; k0 <= qt0 + 63; k0 += 64) {
        __syncthreads();
        // stage K [key][dim] via async Global->LDS DMA (16B per lane per op)
        // and V^T [dim][key] through VGPRs (transpose): 512 chunks each.
        #pragma unroll
        for (int t = 0; t < 4; ++t) {
            int c = tid + t * 128;
            int key = c >> 3, d8 = (c & 7) * 8;
            // LDS offset taken as the low 32 bits of the generic address
            unsigned lds_off =
                (unsigned)(unsigned long long)(uintptr_t)&Kt[key][d8];
            unsigned long long gaddr = (unsigned long long)(uintptr_t)
                &Kb[(size_t)(k0 + key) * HEAD_DIM + d8];
            asm volatile("global_load_async_to_lds_b128 %0, %1, off"
                         :
                         : "v"(lds_off), "v"(gaddr)
                         : "memory");
            v8h vv = *(const v8h*)&Vb[(size_t)(k0 + key) * HEAD_DIM + d8];
            #pragma unroll
            for (int j = 0; j < 8; ++j) Vt[d8 + j][key] = vv[j];
        }
        // own wave's async chunks landed in LDS, then workgroup barrier
        asm volatile("s_wait_asynccnt 0" ::: "memory");
        __syncthreads();

        // ---- S = Q K^T, scaled + causal mask, kept per-lane in C layout ----
        float p[4][8];
        float tmax[8], tsum[8];
        #pragma unroll
        for (int r = 0; r < 8; ++r) tmax[r] = -1e30f;

        #pragma unroll
        for (int nt = 0; nt < 4; ++nt) {
            v16h b0, b1;
            const _Float16* kr = &Kt[nt * 16 + bn][0];
            #pragma unroll
            for (int i = 0; i < 16; ++i) {
                b0[i] = kr[(i < 8 ? kb : 16 + kb) + (i & 7)];
                b1[i] = kr[32 + (i < 8 ? kb : 16 + kb) + (i & 7)];
            }
            v8f s = {};
            s = __builtin_amdgcn_wmma_f32_16x16x32_f16(
                false, aq0, false, b0, (short)0, s, false, false);
            s = __builtin_amdgcn_wmma_f32_16x16x32_f16(
                false, aq1, false, b1, (short)0, s, false, false);

            int key = k0 + nt * 16 + bn;
            #pragma unroll
            for (int r = 0; r < 8; ++r) {
                int q = q0 + mb + r;
                float sc = s[r] * scale;
                if (key > q) sc = -1e30f;
                p[nt][r] = sc;
                tmax[r] = fmaxf(tmax[r], sc);
            }
        }
        // row max across the 16 lanes holding each row (xor stays in half)
        #pragma unroll
        for (int r = 0; r < 8; ++r) {
            float v = tmax[r];
            v = fmaxf(v, __shfl_xor(v, 1));
            v = fmaxf(v, __shfl_xor(v, 2));
            v = fmaxf(v, __shfl_xor(v, 4));
            v = fmaxf(v, __shfl_xor(v, 8));
            tmax[r] = v;
        }
        float corr[8];
        #pragma unroll
        for (int r = 0; r < 8; ++r) {
            float mnew = fmaxf(mrow[r], tmax[r]);
            corr[r] = __expf(mrow[r] - mnew);
            mrow[r] = mnew;
            tsum[r] = 0.f;
        }
        // exponentiate, write P tile to LDS (C layout -> memory)
        #pragma unroll
        for (int nt = 0; nt < 4; ++nt) {
            #pragma unroll
            for (int r = 0; r < 8; ++r) {
                float pe = __expf(p[nt][r] - mrow[r]);
                tsum[r] += pe;
                Pt[wave][mb + r][nt * 16 + bn] = (_Float16)pe;
            }
        }
        #pragma unroll
        for (int r = 0; r < 8; ++r) {
            float v = tsum[r];
            v += __shfl_xor(v, 1);
            v += __shfl_xor(v, 2);
            v += __shfl_xor(v, 4);
            v += __shfl_xor(v, 8);
            lrow[r] = lrow[r] * corr[r] + v;
        }
        // rescale running output
        #pragma unroll
        for (int nt = 0; nt < 4; ++nt)
            #pragma unroll
            for (int r = 0; r < 8; ++r) o[nt][r] *= corr[r];

        // ---- O += P V : reload P from LDS as A fragments ----
        v16h ap0, ap1;
        {
            const _Float16* pr = &Pt[wave][am][0];
            #pragma unroll
            for (int i = 0; i < 8; ++i) {
                ap0[i]     = pr[kb + i];
                ap0[8 + i] = pr[16 + kb + i];
                ap1[i]     = pr[32 + kb + i];
                ap1[8 + i] = pr[48 + kb + i];
            }
        }
        #pragma unroll
        for (int nt = 0; nt < 4; ++nt) {
            v16h bv0, bv1;
            const _Float16* vr = &Vt[nt * 16 + bn][0];
            #pragma unroll
            for (int i = 0; i < 16; ++i) {
                bv0[i] = vr[(i < 8 ? kb : 16 + kb) + (i & 7)];
                bv1[i] = vr[32 + (i < 8 ? kb : 16 + kb) + (i & 7)];
            }
            o[nt] = __builtin_amdgcn_wmma_f32_16x16x32_f16(
                false, ap0, false, bv0, (short)0, o[nt], false, false);
            o[nt] = __builtin_amdgcn_wmma_f32_16x16x32_f16(
                false, ap1, false, bv1, (short)0, o[nt], false, false);
        }
    }

    // epilogue: normalize rows, store to ctx [b*L + q][D_MODEL] (f16)
    const int h_ = bh & (N_HEADS - 1);
    const int b_ = bh >> 4;
    #pragma unroll
    for (int nt = 0; nt < 4; ++nt) {
        int d = nt * 16 + bn;
        #pragma unroll
        for (int r = 0; r < 8; ++r) {
            int q = q0 + mb + r;
            float v = o[nt][r] / lrow[r];
            ctx[((size_t)(b_ * SEQ + q)) * D_MODEL + h_ * HEAD_DIM + d] =
                (_Float16)v;
        }
    }
}

// ---------------------------------------------------------------------------
extern "C" void kernel_launch(void* const* d_in, const int* in_sizes, int n_in,
                              void* d_out, int out_size, void* d_ws, size_t ws_size,
                              hipStream_t stream) {
    const float* hs = (const float*)d_in[0];
    const float* Wq = (const float*)d_in[1];
    const float* bq = (const float*)d_in[2];
    const float* Wk = (const float*)d_in[3];
    const float* bk = (const float*)d_in[4];
    const float* Wv = (const float*)d_in[5];
    const float* bv = (const float*)d_in[6];
    const float* Wo = (const float*)d_in[7];
    const float* bo = (const float*)d_in[8];
    float* out = (float*)d_out;

    // workspace layout (f16 elements): 24M elems = 48 MB total
    _Float16* ws = (_Float16*)d_ws;
    const size_t E_H = (size_t)M_TOTAL * D_MODEL;   // 4M
    const size_t E_W = (size_t)D_MODEL * D_MODEL;   // 1M
    _Float16* h16  = ws;
    _Float16* wq16 = h16 + E_H;
    _Float16* wk16 = wq16 + E_W;
    _Float16* wv16 = wk16 + E_W;
    _Float16* wo16 = wv16 + E_W;
    _Float16* q16  = wo16 + E_W;
    _Float16* k16  = q16 + E_H;
    _Float16* v16  = k16 + E_H;
    _Float16* c16  = v16 + E_H;

    f32_to_f16_kernel<<<1024, 256, 0, stream>>>(hs, h16, (int)E_H);
    f32_to_f16_kernel<<<512, 256, 0, stream>>>(Wq, wq16, (int)E_W);
    f32_to_f16_kernel<<<512, 256, 0, stream>>>(Wk, wk16, (int)E_W);
    f32_to_f16_kernel<<<512, 256, 0, stream>>>(Wv, wv16, (int)E_W);
    f32_to_f16_kernel<<<512, 256, 0, stream>>>(Wo, wo16, (int)E_W);

    dim3 gg(D_MODEL / 64, M_TOTAL / 128);  // (16, 32)
    gemm_wmma<1><<<gg, 128, 0, stream>>>(h16, wq16, bq, nullptr, q16,
                                         M_TOTAL, D_MODEL, D_MODEL);
    gemm_wmma<1><<<gg, 128, 0, stream>>>(h16, wk16, bk, nullptr, k16,
                                         M_TOTAL, D_MODEL, D_MODEL);
    gemm_wmma<1><<<gg, 128, 0, stream>>>(h16, wv16, bv, nullptr, v16,
                                         M_TOTAL, D_MODEL, D_MODEL);

    dim3 ga(SEQ / 64, BATCH * N_HEADS);    // (32, 32)
    attn_wmma<<<ga, 128, 0, stream>>>(q16, k16, v16, c16);

    gemm_wmma<0><<<gg, 128, 0, stream>>>(c16, wo16, bo, out, nullptr,
                                         M_TOTAL, D_MODEL, D_MODEL);
}
